// UNetSCN_71579924955585
// MI455X (gfx1250) — matrix operand order repaired and checked by reading, not compiled, not run on hardware
//
#include <hip/hip_runtime.h>

typedef __bf16 bf16;
typedef __attribute__((ext_vector_type(16))) __bf16 v16bf;
typedef __attribute__((ext_vector_type(8)))  __bf16 v8bf;
typedef __attribute__((ext_vector_type(8)))  float  v8f;

union AFrag { v16bf v; v8bf h[2]; };

// ---------------------------------------------------------------------------
// Elementwise helpers
// ---------------------------------------------------------------------------
__global__ void k_mask_init(const int* __restrict__ mi, float* __restrict__ mf, long n) {
  long i = (long)blockIdx.x * blockDim.x + threadIdx.x;
  if (i < n) mf[i] = (float)mi[i];
}

// coarse[b,z,y,x] = max over 2x2x2 window of fine
__global__ void k_mask_down(const float* __restrict__ fine, float* __restrict__ coarse,
                            int Gc, long nvc) {
  long i = (long)blockIdx.x * blockDim.x + threadIdx.x;
  if (i >= nvc) return;
  int x = (int)(i % Gc); long t = i / Gc;
  int y = (int)(t % Gc); t /= Gc;
  int z = (int)(t % Gc); int b = (int)(t / Gc);
  int Gf = Gc * 2;
  float m = 0.f;
  for (int dz = 0; dz < 2; ++dz)
    for (int dy = 0; dy < 2; ++dy)
      for (int dx = 0; dx < 2; ++dx) {
        long idx = (((long)b * Gf + (2 * z + dz)) * Gf + (2 * y + dy)) * Gf + (2 * x + dx);
        m = fmaxf(m, fine[idx]);
      }
  coarse[i] = m;
}

// stem: 3^3 conv, Cin=1 -> Cout=16, input premasked, output masked; channel-last out
__global__ void k_stem(const float* __restrict__ feats, const float* __restrict__ mask,
                       const float* __restrict__ w, float* __restrict__ out,
                       int G, long nv) {
  long v = (long)blockIdx.x * blockDim.x + threadIdx.x;
  if (v >= nv) return;
  int x = (int)(v % G); long t = v / G;
  int y = (int)(t % G); t /= G;
  int z = (int)(t % G); int b = (int)(t / G);
  float acc[16];
#pragma unroll
  for (int c = 0; c < 16; ++c) acc[c] = 0.f;
  for (int tz = 0; tz < 3; ++tz) {
    int zi = z - 1 + tz; if ((unsigned)zi >= (unsigned)G) continue;
    for (int ty = 0; ty < 3; ++ty) {
      int yi = y - 1 + ty; if ((unsigned)yi >= (unsigned)G) continue;
      for (int tx = 0; tx < 3; ++tx) {
        int xi = x - 1 + tx; if ((unsigned)xi >= (unsigned)G) continue;
        long idx = (((long)b * G + zi) * G + yi) * G + xi;
        float f = feats[idx] * mask[idx];
        const float* wt = w + ((tz * 3 + ty) * 3 + tx) * 16;
#pragma unroll
        for (int c = 0; c < 16; ++c) acc[c] += f * wt[c];
      }
    }
  }
  float mk = mask[v];
#pragma unroll
  for (int c = 0; c < 16; ++c) out[v * 16 + c] = acc[c] * mk;
}

// fused BN + ReLU + mask, f32 (row stride Cs) -> bf16 (row stride Cd, offset cofs).
// Writes Cw channels per voxel; channels >= C are zero-filled (K padding for WMMA).
__global__ void k_bnrelu(const float* __restrict__ src, int Cs,
                         const float* __restrict__ g, const float* __restrict__ bt,
                         const float* __restrict__ mask,
                         bf16* __restrict__ dst, int Cd, int cofs, int C, int Cw, long nvox) {
  long i = (long)blockIdx.x * blockDim.x + threadIdx.x;
  long total = nvox * Cw;
  if (i >= total) return;
  int c = (int)(i % Cw); long v = i / Cw;
  float val = 0.f;
  if (c < C) {
    val = fmaf(src[v * (long)Cs + c], g[c], bt[c]);
    val = fmaxf(val, 0.f) * mask[v];
  }
  dst[v * (long)Cd + cofs + c] = (bf16)val;
}

// weight transpose + K-pad: src[tap][k][n] (f32) -> dst[tap][n][kpad] (bf16, zero pad)
__global__ void k_w_transpose(const float* __restrict__ src, bf16* __restrict__ dst,
                              int Cin, int Cout, int Kpad, long total) {
  long i = (long)blockIdx.x * blockDim.x + threadIdx.x;
  if (i >= total) return;
  int k = (int)(i % Kpad); long t = i / Kpad;
  int n = (int)(t % Cout); int tap = (int)(t / Cout);
  dst[i] = (k < Cin) ? (bf16)src[((long)tap * Cin + k) * Cout + n] : (bf16)0.f;
}

// final BN+ReLU+mask, channel-last f32 -> NCDHW f32 output
__global__ void k_final(const float* __restrict__ x, const float* __restrict__ g,
                        const float* __restrict__ bt, const float* __restrict__ mask,
                        float* __restrict__ out, int G, long nv) {
  long i = (long)blockIdx.x * blockDim.x + threadIdx.x;
  long total = nv * 16;
  if (i >= total) return;
  int c = (int)(i % 16); long v = i / 16;
  int xx = (int)(v % G); long t = v / G;
  int yy = (int)(t % G); t /= G;
  int zz = (int)(t % G); int b = (int)(t / G);
  float val = fmaxf(fmaf(x[v * 16 + c], g[c], bt[c]), 0.f) * mask[v];
  out[((((long)b * 16 + c) * G + zz) * G + yy) * G + xx] = val;
}

// ---------------------------------------------------------------------------
// Implicit-GEMM conv via v_wmma_f32_16x16x32_bf16.
// Wave = 16 output voxels x 16 output channels; K = taps * CINPAD, fully
// unrolled in 32-deep chunks (CINPAD is the zero-padded channel stride).
// Fragment loads use contiguous k order (k = 16*(L>>4) + s) for A and B:
//   A: lane L -> row m=L&15, needs K = kbA..kbA+15
//      (kbA=(L>>4)*16) -> two contiguous global_load_b128 from channel-last rows.
//   B: lane L -> col n=L&15, needs K = kbB..kbB+15 (kbB=(L>>4)*16)
//      -> one contiguous 32B load from the pre-transposed [tap][n][Kpad] weights.
//   C/D: lane L -> col n=L&15, rows m=(L>>4)*8+v.
// ---------------------------------------------------------------------------
template <int CINPAD, int KSZ>
__global__ __launch_bounds__(256)
void k_conv_wmma(const bf16* __restrict__ in, const bf16* __restrict__ wgtT,
                 const float* __restrict__ mask, float* __restrict__ out,
                 int Din, int Hin, int Win,
                 int Dout, int Hout, int Wout,
                 int Cout, int stride, int pad, int nvoxOut) {
  const int lane = threadIdx.x & 31;
  const int wv   = threadIdx.x >> 5;
  const int voxBase = (blockIdx.x * 8 + wv) * 16;
  if (voxBase >= nvoxOut) return;          // wave-uniform: EXEC stays all-1s
  const int n0  = blockIdx.y * 16;
  const int nB  = lane & 15;
  const int kbA = (lane >> 4) * 16;
  const int kbB = (lane >> 4) * 16;

  int vA = voxBase + (lane & 15);
  bool rowValid = vA < nvoxOut;
  int va = rowValid ? vA : (nvoxOut - 1);
  int ox = va % Wout; int t = va / Wout;
  int oy = t % Hout;  t /= Hout;
  int oz = t % Dout;  int ob = t / Dout;

  v8bf zero;
#pragma unroll
  for (int e = 0; e < 8; ++e) zero[e] = (bf16)0.f;

  v8f acc = {0.f, 0.f, 0.f, 0.f, 0.f, 0.f, 0.f, 0.f};
  const bf16* wrow = wgtT + (long)(n0 + nB) * CINPAD;  // this lane's B row (tap 0)

  for (int tz = 0; tz < KSZ; ++tz) {
    int zi = oz * stride - pad + tz;
    for (int ty = 0; ty < KSZ; ++ty) {
      int yi = oy * stride - pad + ty;
      for (int tx = 0; tx < KSZ; ++tx) {
        int xi = ox * stride - pad + tx;
        bool vv = rowValid && (unsigned)zi < (unsigned)Din &&
                  (unsigned)yi < (unsigned)Hin && (unsigned)xi < (unsigned)Win;
        const bf16* arow =
            in + ((long)(((ob * Din + zi) * Hin + yi) * Win + xi)) * CINPAD;
        const bf16* wtap =
            wrow + (long)((tz * KSZ + ty) * KSZ + tx) * Cout * CINPAD;
#pragma unroll
        for (int c0 = 0; c0 < CINPAD; c0 += 32) {
          AFrag a;
          a.h[0] = vv ? *(const v8bf*)(arow + c0 + kbA)      : zero;
          a.h[1] = vv ? *(const v8bf*)(arow + c0 + kbA + 8) : zero;
          v16bf bm = *(const v16bf*)(wtap + c0 + kbB);
          acc = __builtin_amdgcn_wmma_f32_16x16x32_bf16(false, a.v, false, bm,
                                                        (short)0, acc, false, false);
        }
      }
    }
  }

  const int mBase = (lane >> 4) * 8;
#pragma unroll
  for (int v = 0; v < 8; ++v) {
    int vox = voxBase + mBase + v;
    if (vox < nvoxOut)
      out[(long)vox * Cout + n0 + nB] = acc[v] * mask[vox];
  }
}

// ---------------------------------------------------------------------------
// k=2 s=2 transposed conv as 8 parity-indexed 1x1 GEMMs over the coarse grid.
// Output parity (pz,py,px) uses kernel tap (1-pz,1-py,1-px)
// (jax.lax.conv_transpose, transpose_kernel=False).
// ---------------------------------------------------------------------------
template <int CINPAD>
__global__ __launch_bounds__(256)
void k_deconv_wmma(const bf16* __restrict__ in, const bf16* __restrict__ wgtT,
                   const float* __restrict__ maskF, float* __restrict__ outF,
                   int Gc, int Cout, int nvoxC) {
  const int lane = threadIdx.x & 31;
  const int wv   = threadIdx.x >> 5;
  const int voxBase = (blockIdx.x * 8 + wv) * 16;
  if (voxBase >= nvoxC) return;
  const int n0  = blockIdx.y * 16;
  const int nB  = lane & 15;
  const int kbA = (lane >> 4) * 16;
  const int kbB = (lane >> 4) * 16;
  const int p  = blockIdx.z;
  const int pz = (p >> 2) & 1, py = (p >> 1) & 1, px = p & 1;
  const int tap = ((1 - pz) * 2 + (1 - py)) * 2 + (1 - px);

  int vA = voxBase + (lane & 15);
  bool rowValid = vA < nvoxC;
  const bf16* arow = in + (long)(rowValid ? vA : 0) * CINPAD;
  const bf16* wrow = wgtT + ((long)tap * Cout + n0 + nB) * CINPAD;

  v8bf zero;
#pragma unroll
  for (int e = 0; e < 8; ++e) zero[e] = (bf16)0.f;

  v8f acc = {0.f, 0.f, 0.f, 0.f, 0.f, 0.f, 0.f, 0.f};
#pragma unroll
  for (int c0 = 0; c0 < CINPAD; c0 += 32) {
    AFrag a;
    a.h[0] = rowValid ? *(const v8bf*)(arow + c0 + kbA)      : zero;
    a.h[1] = rowValid ? *(const v8bf*)(arow + c0 + kbA + 8) : zero;
    v16bf bm = *(const v16bf*)(wrow + c0 + kbB);
    acc = __builtin_amdgcn_wmma_f32_16x16x32_bf16(false, a.v, false, bm,
                                                  (short)0, acc, false, false);
  }

  const int Gf = Gc * 2;
  const int mBase = (lane >> 4) * 8;
#pragma unroll
  for (int v = 0; v < 8; ++v) {
    int vox = voxBase + mBase + v;
    if (vox < nvoxC) {
      int xx = vox % Gc; int t = vox / Gc;
      int yy = t % Gc;   t /= Gc;
      int zz = t % Gc;   int bb = t / Gc;
      long fidx = (((long)bb * Gf + (2 * zz + pz)) * Gf + (2 * yy + py)) * Gf + (2 * xx + px);
      outF[fidx * Cout + n0 + nB] = acc[v] * maskF[fidx];
    }
  }
}

// ---------------------------------------------------------------------------
// Host orchestration: flattened U-Net recursion (down sweep + up sweep)
// ---------------------------------------------------------------------------
static const int PLANES_H[7] = {16, 32, 48, 64, 80, 96, 112};

extern "C" void kernel_launch(void* const* d_in, const int* in_sizes, int n_in,
                              void* d_out, int out_size, void* d_ws, size_t ws_size,
                              hipStream_t stream) {
  (void)in_sizes; (void)n_in; (void)out_size; (void)ws_size;
  const int B = 2, G0 = 64;
  auto rup = [](int x) { return (x + 31) & ~31; };

  // --- input pointers (setup_inputs dict insertion order, nested in order) ---
  int idx = 0;
  const float* feats   = (const float*)d_in[idx++];
  const int*   mask_i  = (const int*)  d_in[idx++];
  const float* stem_wf = (const float*)d_in[idx++];
  const float* fin_g   = (const float*)d_in[idx++];
  const float* fin_b   = (const float*)d_in[idx++];
  const float *pre_g[7], *pre_b[7], *pre_wf[7];
  const float *down_g[6], *down_b[6], *down_wf[6];
  const float *up_g[6], *up_b[6], *up_wf[6];
  const float *post_g[6], *post_b[6], *post_wf[6];
  for (int l = 0; l < 7; ++l) { pre_g[l] = (const float*)d_in[idx++]; pre_b[l] = (const float*)d_in[idx++]; }
  for (int l = 0; l < 7; ++l) pre_wf[l] = (const float*)d_in[idx++];
  for (int l = 0; l < 6; ++l) { down_g[l] = (const float*)d_in[idx++]; down_b[l] = (const float*)d_in[idx++]; }
  for (int l = 0; l < 6; ++l) down_wf[l] = (const float*)d_in[idx++];
  for (int l = 0; l < 6; ++l) { up_g[l] = (const float*)d_in[idx++]; up_b[l] = (const float*)d_in[idx++]; }
  for (int l = 0; l < 6; ++l) up_wf[l] = (const float*)d_in[idx++];
  for (int l = 0; l < 6; ++l) { post_g[l] = (const float*)d_in[idx++]; post_b[l] = (const float*)d_in[idx++]; }
  for (int l = 0; l < 6; ++l) post_wf[l] = (const float*)d_in[idx++];

  // --- workspace bump allocator ---
  char* base = (char*)d_ws;
  size_t off = 0;
  auto alloc = [&](size_t bytes) -> void* {
    off = (off + 255) & ~(size_t)255;
    void* p = base + off;
    off += bytes;
    return p;
  };

  int Gl[7]; long nv[7];
  float *maskf[7], *xf[7], *yf[6];
  bf16 *act[7], *wpre[7], *wpost[6], *wdown[6], *wup[6];
  for (int l = 0; l < 7; ++l) {
    Gl[l] = G0 >> l;
    nv[l] = (long)B * Gl[l] * Gl[l] * Gl[l];
    int c = PLANES_H[l];
    maskf[l] = (float*)alloc((size_t)nv[l] * 4);
    xf[l]    = (float*)alloc((size_t)nv[l] * c * 4);
    act[l]   = (bf16*) alloc((size_t)nv[l] * 2 * c * 2);      // 2c >= rup(c,32)
    wpre[l]  = (bf16*) alloc((size_t)27 * c * rup(c) * 2);    // [tap][cout=c][Kpad]
    if (l < 6) {
      int c2 = PLANES_H[l + 1];
      yf[l]    = (float*)alloc((size_t)nv[l] * c * 4);
      wpost[l] = (bf16*) alloc((size_t)27 * c * (2 * c) * 2); // Kpad = 2c (mult of 32)
      wdown[l] = (bf16*) alloc((size_t)8 * c2 * rup(c) * 2);
      wup[l]   = (bf16*) alloc((size_t)8 * c * rup(c2) * 2);
    }
  }

  auto ew = [](long n) { return dim3((unsigned)((n + 255) / 256)); };
  auto wT = [&](const float* s, bf16* d, int taps, int Cin, int Cout, int Kpad) {
    long total = (long)taps * Cout * Kpad;
    k_w_transpose<<<ew(total), 256, 0, stream>>>(s, d, Cin, Cout, Kpad, total);
  };
  auto conv = [&](const bf16* in, const bf16* w, const float* mk, float* out,
                  int Gin, int Gout, int CinPad, int Cout, int ksz, int stride, int pad) {
    int nvo = B * Gout * Gout * Gout;
    int mt = (nvo + 15) / 16;
    dim3 grid((unsigned)((mt + 7) / 8), (unsigned)(Cout / 16), 1);
#define LC(CP, KS) k_conv_wmma<CP, KS><<<grid, 256, 0, stream>>>( \
    in, w, mk, out, Gin, Gin, Gin, Gout, Gout, Gout, Cout, stride, pad, nvo)
    if (ksz == 3) {
      switch (CinPad) {
        case 32:  LC(32, 3);  break;
        case 64:  LC(64, 3);  break;
        case 96:  LC(96, 3);  break;
        case 128: LC(128, 3); break;
        case 160: LC(160, 3); break;
        case 192: LC(192, 3); break;
      }
    } else {
      switch (CinPad) {
        case 32:  LC(32, 2);  break;
        case 64:  LC(64, 2);  break;
        case 96:  LC(96, 2);  break;
        case 128: LC(128, 2); break;
      }
    }
#undef LC
  };
  auto deconv = [&](const bf16* in, const bf16* w, const float* mkF, float* outF,
                    int Gc, int CinPad, int Cout) {
    int nvc = B * Gc * Gc * Gc;
    int mt = (nvc + 15) / 16;
    dim3 grid((unsigned)((mt + 7) / 8), (unsigned)(Cout / 16), 8);
#define LD(CP) k_deconv_wmma<CP><<<grid, 256, 0, stream>>>(in, w, mkF, outF, Gc, Cout, nvc)
    switch (CinPad) {
      case 32:  LD(32);  break;
      case 64:  LD(64);  break;
      case 96:  LD(96);  break;
      case 128: LD(128); break;
    }
#undef LD
  };

  // --- weight transpose + pad (f32 [tap][k][n] -> bf16 [tap][n][Kpad]) ---
  for (int l = 0; l < 7; ++l) {
    int c = PLANES_H[l];
    wT(pre_wf[l], wpre[l], 27, c, c, rup(c));
    if (l < 6) {
      int c2 = PLANES_H[l + 1];
      wT(post_wf[l], wpost[l], 27, 2 * c, c, 2 * c);
      wT(down_wf[l], wdown[l], 8, c, c2, rup(c));
      wT(up_wf[l],   wup[l],   8, c2, c, rup(c2));
    }
  }

  // --- active-set mask hierarchy ---
  k_mask_init<<<ew(nv[0]), 256, 0, stream>>>(mask_i, maskf[0], nv[0]);
  for (int l = 0; l < 6; ++l)
    k_mask_down<<<ew(nv[l + 1]), 256, 0, stream>>>(maskf[l], maskf[l + 1], Gl[l + 1], nv[l + 1]);

  // --- stem: 3^3 conv 1->16 (channel-last f32 into xf[0]) ---
  k_stem<<<ew(nv[0]), 256, 0, stream>>>(feats, maskf[0], stem_wf, xf[0], G0, nv[0]);

  // --- down sweep ---
  for (int l = 0; l < 7; ++l) {
    int c = PLANES_H[l], cp = rup(c);
    k_bnrelu<<<ew(nv[l] * cp), 256, 0, stream>>>(xf[l], c, pre_g[l], pre_b[l], maskf[l],
                                                 act[l], cp, 0, c, cp, nv[l]);
    conv(act[l], wpre[l], maskf[l], xf[l], Gl[l], Gl[l], cp, c, 3, 1, 1);
    if (l < 6) {
      int c2 = PLANES_H[l + 1];
      k_bnrelu<<<ew(nv[l] * cp), 256, 0, stream>>>(xf[l], c, down_g[l], down_b[l], maskf[l],
                                                   act[l], cp, 0, c, cp, nv[l]);
      conv(act[l], wdown[l], maskf[l + 1], xf[l + 1], Gl[l], Gl[l + 1], cp, c2, 2, 2, 0);
    }
  }

  // --- up sweep ---
  for (int l = 5; l >= 0; --l) {
    int c = PLANES_H[l], c2 = PLANES_H[l + 1], c2p = rup(c2);
    k_bnrelu<<<ew(nv[l + 1] * c2p), 256, 0, stream>>>(xf[l + 1], c2, up_g[l], up_b[l],
                                                      maskf[l + 1], act[l + 1], c2p, 0, c2, c2p,
                                                      nv[l + 1]);
    deconv(act[l + 1], wup[l], maskf[l], yf[l], Gl[l + 1], c2p, c);
    // JoinTable: bnrelu x -> channels [0,c), bnrelu y -> channels [c,2c); 2c % 32 == 0
    k_bnrelu<<<ew(nv[l] * c), 256, 0, stream>>>(xf[l], c, post_g[l], post_b[l], maskf[l],
                                                act[l], 2 * c, 0, c, c, nv[l]);
    k_bnrelu<<<ew(nv[l] * c), 256, 0, stream>>>(yf[l], c, post_g[l] + c, post_b[l] + c, maskf[l],
                                                act[l], 2 * c, c, c, c, nv[l]);
    conv(act[l], wpost[l], maskf[l], xf[l], Gl[l], Gl[l], 2 * c, c, 3, 1, 1);
  }

  // --- final BN+ReLU, channel-last -> NCDHW f32 output ---
  k_final<<<ew(nv[0] * 16), 256, 0, stream>>>(xf[0], fin_g, fin_b, maskf[0],
                                              (float*)d_out, G0, nv[0]);
}
